// ParallelHeteroGNN_41343355191517
// MI455X (gfx1250) — matrix operand order repaired and checked by reading, not compiled, not run on hardware
//
#include <hip/hip_runtime.h>
#include <hip/hip_bf16.h>

#define GN 100000
#define GIN 256
#define GF 128        // HEADS*HID
#define GOUT 64
#define GE 1600000
#define GEN (GE + GN) // edges + self loops
#define NEG_SLOPE 0.2f

typedef __bf16 v16bf __attribute__((ext_vector_type(16)));
typedef float  v8f   __attribute__((ext_vector_type(8)));

__device__ __forceinline__ unsigned fenc(float f) {
    unsigned u = __float_as_uint(f);
    return (u & 0x80000000u) ? ~u : (u | 0x80000000u);
}
__device__ __forceinline__ float fdec(unsigned k) {
    unsigned u = (k & 0x80000000u) ? (k & 0x7FFFFFFFu) : ~k;
    return __uint_as_float(u);
}
__device__ __forceinline__ float lrelu(float v) { return v > 0.f ? v : NEG_SLOPE * v; }

// ---------------- K1: fused h1 = x@W1, h2 = x@W2 (bf16 WMMA, f32 acc) ----------------
// grid.x = N/16, block = 256 (8 waves); wave w owns output cols [16w,16w+16)
__global__ __launch_bounds__(256) void k_gemm_h(
    const float* __restrict__ x, const float* __restrict__ W1,
    const float* __restrict__ W2, float* __restrict__ hbase)
{
    __shared__ float xs[16 * 257];
    const int rb = blockIdx.x * 16;
    const int t  = threadIdx.x;
#pragma unroll
    for (int i = 0; i < 16; ++i) {
        int f = t + i * 256;
        int row = f >> 8, col = f & 255;
        xs[row * 257 + col] = x[(size_t)(rb + row) * GIN + col];
    }
    __syncthreads();
    const int wave = t >> 5, lane = t & 31;
    const int hi = lane >> 4, mn = lane & 15;
    const int n0 = wave * 16;
    v8f acc1 = {0.f, 0.f, 0.f, 0.f, 0.f, 0.f, 0.f, 0.f};
    v8f acc2 = acc1;
#pragma unroll
    for (int k0 = 0; k0 < GIN; k0 += 32) {
        v16bf a, b1, b2;
#pragma unroll
        for (int e = 0; e < 16; ++e) {
            int ka = (e < 8) ? (8 * hi + e) : (16 + 8 * hi + (e - 8));
            a[e] = (__bf16)xs[mn * 257 + k0 + ka];
            int kb = (e < 8) ? (8 * hi + e) : (16 + 8 * hi + (e - 8));
            b1[e] = (__bf16)W1[(size_t)(k0 + kb) * GF + n0 + mn];
            b2[e] = (__bf16)W2[(size_t)(k0 + kb) * GF + n0 + mn];
        }
        acc1 = __builtin_amdgcn_wmma_f32_16x16x32_bf16(false, a, false, b1, (short)0, acc1, false, false);
        acc2 = __builtin_amdgcn_wmma_f32_16x16x32_bf16(false, a, false, b2, (short)0, acc2, false, false);
    }
    float* h1 = hbase;
    float* h2 = hbase + (size_t)GN * GF;
#pragma unroll
    for (int r = 0; r < 8; ++r) {
        size_t off = (size_t)(rb + r + 8 * hi) * GF + n0 + mn;
        h1[off] = acc1[r];
        h2[off] = acc2[r];
    }
}

// ---------------- K2: per-(node,head) attention coefficients ----------------
__global__ __launch_bounds__(256) void k_att(
    const float* __restrict__ hbase,
    const float* __restrict__ as1, const float* __restrict__ ad1,
    const float* __restrict__ as2, const float* __restrict__ ad2,
    float* __restrict__ a_src, float* __restrict__ a_dst)
{
    const int g = blockIdx.y;
    const int idx = blockIdx.x * blockDim.x + threadIdx.x; // node*2+head
    if (idx >= GN * 2) return;
    const int node = idx >> 1, head = idx & 1;
    const float* hrow = hbase + (size_t)g * GN * GF + (size_t)node * GF + head * 64;
    const float* asv = (g ? as2 : as1) + head * 64;
    const float* adv = (g ? ad2 : ad1) + head * 64;
    float s = 0.f, d = 0.f;
#pragma unroll 4
    for (int c = 0; c < 64; ++c) { float v = hrow[c]; s += v * asv[c]; d += v * adv[c]; }
    a_src[(size_t)g * GN * 2 + idx] = s;
    a_dst[(size_t)g * GN * 2 + idx] = d;
}

// ---------------- K3: segment max (order-preserving uint atomicMax) ----------------
__global__ __launch_bounds__(256) void k_edge_max(
    const int* __restrict__ ei_c, const int* __restrict__ ei_l,
    const float* __restrict__ a_src, const float* __restrict__ a_dst,
    unsigned* __restrict__ m)
{
    const int g = blockIdx.y;
    const int i = blockIdx.x * blockDim.x + threadIdx.x;
    if (i >= GEN) return;
    const int* ei = g ? ei_l : ei_c;
    int s, d;
    if (i < GE) { s = ei[i]; d = ei[GE + i]; } else { s = d = i - GE; }
    const float* as = a_src + (size_t)g * GN * 2;
    const float* ad = a_dst + (size_t)g * GN * 2;
    unsigned* mg = m + (size_t)g * GN * 2;
#pragma unroll
    for (int h = 0; h < 2; ++h) {
        float e = lrelu(as[s * 2 + h] + ad[d * 2 + h]);
        atomicMax(&mg[d * 2 + h], fenc(e));
    }
}

// ---------------- K4: segment sum of exp(e - max) ----------------
__global__ __launch_bounds__(256) void k_edge_sum(
    const int* __restrict__ ei_c, const int* __restrict__ ei_l,
    const float* __restrict__ a_src, const float* __restrict__ a_dst,
    const unsigned* __restrict__ m, float* __restrict__ denom)
{
    const int g = blockIdx.y;
    const int i = blockIdx.x * blockDim.x + threadIdx.x;
    if (i >= GEN) return;
    const int* ei = g ? ei_l : ei_c;
    int s, d;
    if (i < GE) { s = ei[i]; d = ei[GE + i]; } else { s = d = i - GE; }
    const float* as = a_src + (size_t)g * GN * 2;
    const float* ad = a_dst + (size_t)g * GN * 2;
    const unsigned* mg = m + (size_t)g * GN * 2;
    float* dg = denom + (size_t)g * GN * 2;
#pragma unroll
    for (int h = 0; h < 2; ++h) {
        float e = lrelu(as[s * 2 + h] + ad[d * 2 + h]);
        atomicAdd(&dg[d * 2 + h], __expf(e - fdec(mg[d * 2 + h])));
    }
}

// ---------------- K5: weighted scatter-add, one wave per edge ----------------
// block = 256 (8 waves); lane c owns channels [4c,4c+4); head = lane>>4
__global__ __launch_bounds__(256) void k_edge_aggr(
    const int* __restrict__ ei_c, const int* __restrict__ ei_l,
    const float* __restrict__ a_src, const float* __restrict__ a_dst,
    const unsigned* __restrict__ m, const float* __restrict__ denom,
    const float* __restrict__ hbase, float* __restrict__ obase)
{
    const int g = blockIdx.y;
    const int lane = threadIdx.x & 31;
    const int i = blockIdx.x * (blockDim.x >> 5) + (threadIdx.x >> 5);
    if (i >= GEN) return;
    const int* ei = g ? ei_l : ei_c;
    int s, d;
    if (i < GE) { s = ei[i]; d = ei[GE + i]; } else { s = d = i - GE; }
    const int head = lane >> 4;
    const float* as = a_src + (size_t)g * GN * 2;
    const float* ad = a_dst + (size_t)g * GN * 2;
    const unsigned* mg = m + (size_t)g * GN * 2;
    const float* dg = denom + (size_t)g * GN * 2;
    float e = lrelu(as[s * 2 + head] + ad[d * 2 + head]);
    float alpha = __expf(e - fdec(mg[d * 2 + head])) / dg[d * 2 + head];
    const float4 hv = *(const float4*)(hbase + (size_t)g * GN * GF + (size_t)s * GF + lane * 4);
    float* orow = obase + (size_t)g * GN * GF + (size_t)d * GF + lane * 4;
    atomicAdd(orow + 0, hv.x * alpha);
    atomicAdd(orow + 1, hv.y * alpha);
    atomicAdd(orow + 2, hv.z * alpha);
    atomicAdd(orow + 3, hv.w * alpha);
}

// ---------------- K6: out = relu(concat(o1+b1, o2+b2)) @ fc_W + fc_b (bf16 WMMA) ----------------
// grid.x = N/16, block = 128 (4 waves of 16 output cols)
__global__ __launch_bounds__(128) void k_fc(
    const float* __restrict__ obase, const float* __restrict__ b1,
    const float* __restrict__ b2, const float* __restrict__ fcW,
    const float* __restrict__ fcb, float* __restrict__ out)
{
    __shared__ float cs[16 * 257];
    const int rb = blockIdx.x * 16;
    const int t  = threadIdx.x;
#pragma unroll
    for (int i = 0; i < 32; ++i) {
        int f = t + i * 128;
        int row = f >> 8, col = f & 255;
        int g = col >> 7, c = col & 127;
        float v = obase[(size_t)g * GN * GF + (size_t)(rb + row) * GF + c] + (g ? b2[c] : b1[c]);
        cs[row * 257 + col] = v > 0.f ? v : 0.f;
    }
    __syncthreads();
    const int wave = t >> 5, lane = t & 31;
    const int hi = lane >> 4, mn = lane & 15;
    const int n0 = wave * 16;
    v8f acc = {0.f, 0.f, 0.f, 0.f, 0.f, 0.f, 0.f, 0.f};
#pragma unroll
    for (int k0 = 0; k0 < 256; k0 += 32) {
        v16bf a, b;
#pragma unroll
        for (int e = 0; e < 16; ++e) {
            int ka = (e < 8) ? (8 * hi + e) : (16 + 8 * hi + (e - 8));
            a[e] = (__bf16)cs[mn * 257 + k0 + ka];
            int kb = (e < 8) ? (8 * hi + e) : (16 + 8 * hi + (e - 8));
            b[e] = (__bf16)fcW[(size_t)(k0 + kb) * GOUT + n0 + mn];
        }
        acc = __builtin_amdgcn_wmma_f32_16x16x32_bf16(false, a, false, b, (short)0, acc, false, false);
    }
#pragma unroll
    for (int r = 0; r < 8; ++r)
        out[(size_t)(rb + r + 8 * hi) * GOUT + n0 + mn] = acc[r] + fcb[n0 + mn];
}

extern "C" void kernel_launch(void* const* d_in, const int* in_sizes, int n_in,
                              void* d_out, int out_size, void* d_ws, size_t ws_size,
                              hipStream_t stream) {
    const float* x    = (const float*)d_in[0];
    const int*   ei_c = (const int*)d_in[1];
    const int*   ei_l = (const int*)d_in[2];
    const float* W1   = (const float*)d_in[3];
    const float* as1  = (const float*)d_in[4];
    const float* ad1  = (const float*)d_in[5];
    const float* b1   = (const float*)d_in[6];
    const float* W2   = (const float*)d_in[7];
    const float* as2  = (const float*)d_in[8];
    const float* ad2  = (const float*)d_in[9];
    const float* b2   = (const float*)d_in[10];
    const float* fcW  = (const float*)d_in[11];
    const float* fcb  = (const float*)d_in[12];
    float* out = (float*)d_out;

    // workspace layout (floats): h[2][N][128] | o[2][N][128] | a_src[2][N][2] |
    //                            a_dst[2][N][2] | m(u32)[2][N][2] | denom[2][N][2]
    float*    h    = (float*)d_ws;
    float*    o    = h + (size_t)2 * GN * GF;
    float*    asrc = o + (size_t)2 * GN * GF;
    float*    adst = asrc + (size_t)2 * GN * 2;
    unsigned* m    = (unsigned*)(adst + (size_t)2 * GN * 2);
    float*    den  = (float*)(m + (size_t)2 * GN * 2);

    // zero accumulators each call (graph-capture safe)
    hipMemsetAsync(o,   0, (size_t)2 * GN * GF * sizeof(float), stream);
    hipMemsetAsync(m,   0, (size_t)2 * GN * 2 * sizeof(unsigned), stream); // 0 == key(-inf)
    hipMemsetAsync(den, 0, (size_t)2 * GN * 2 * sizeof(float), stream);

    k_gemm_h<<<dim3(GN / 16), dim3(256), 0, stream>>>(x, W1, W2, h);
    k_att<<<dim3((GN * 2 + 255) / 256, 2), dim3(256), 0, stream>>>(
        h, as1, ad1, as2, ad2, asrc, adst);
    k_edge_max<<<dim3((GEN + 255) / 256, 2), dim3(256), 0, stream>>>(
        ei_c, ei_l, asrc, adst, m);
    k_edge_sum<<<dim3((GEN + 255) / 256, 2), dim3(256), 0, stream>>>(
        ei_c, ei_l, asrc, adst, m, den);
    k_edge_aggr<<<dim3((GEN + 7) / 8, 2), dim3(256), 0, stream>>>(
        ei_c, ei_l, asrc, adst, m, den, h, o);
    k_fc<<<dim3(GN / 16), dim3(128), 0, stream>>>(o, b1, b2, fcW, fcb, out);
}
